// RNN_22522808500717
// MI455X (gfx1250) — matrix operand order repaired and checked by reading, not compiled, not run on hardware
//
#include <hip/hip_runtime.h>
#include <stdint.h>

// Problem constants (from reference)
#define B_    64
#define T_    512
#define INP_  128
#define REC_  1024
#define OUT_  64
#define ALPHA_ 0.1f
#define NSD_   0.1f

typedef float  v8f   __attribute__((ext_vector_type(8)));
typedef float  v4f   __attribute__((ext_vector_type(4)));
typedef __bf16 bf8   __attribute__((ext_vector_type(8)));
typedef __bf16 bf16v __attribute__((ext_vector_type(16)));
typedef unsigned int u32x4 __attribute__((ext_vector_type(4)));
typedef int          i32x4 __attribute__((ext_vector_type(4)));
typedef int          i32x8 __attribute__((ext_vector_type(8)));

static __device__ inline bf16v cat8(bf8 a, bf8 b) {
  return __builtin_shufflevector(a, b, 0,1,2,3,4,5,6,7,8,9,10,11,12,13,14,15);
}

// Split 8 contiguous fp32 into bf16 (hi, lo) with lo = x - float(hi)  (bf16x3 trick)
static __device__ inline void split8(const float* __restrict__ p, bf8& hi, bf8& lo) {
  v4f x0 = *(const v4f*)p;
  v4f x1 = *(const v4f*)(p + 4);
#pragma unroll
  for (int i = 0; i < 4; ++i) {
    float a = x0[i]; __bf16 h = (__bf16)a; hi[i]     = h; lo[i]     = (__bf16)(a - (float)h);
    float b = x1[i]; __bf16 g = (__bf16)b; hi[4 + i] = g; lo[4 + i] = (__bf16)(b - (float)g);
  }
}

// D = (Ah+Al)*(Bh+Bl) + C  dropping the Al*Bl term (~2^-18 relative)
static __device__ inline v8f wmma3(bf16v ah, bf16v al, bf16v bh, bf16v bl, v8f c) {
  c = __builtin_amdgcn_wmma_f32_16x16x32_bf16(false, ah, false, bh, (short)0, c, false, false);
  c = __builtin_amdgcn_wmma_f32_16x16x32_bf16(false, ah, false, bl, (short)0, c, false, false);
  c = __builtin_amdgcn_wmma_f32_16x16x32_bf16(false, al, false, bh, (short)0, c, false, false);
  return c;
}

// A-fragment (16x32 bf16, lane->M, halves hold K 0..15 / 16..31, matching B)
static __device__ inline void loadA_f32(const float* __restrict__ rowp, int kbase, int half,
                                        bf16v& ahi, bf16v& alo) {
  const float* p0 = rowp + kbase + half * 16;
  bf8 h0, l0, h1, l1;
  split8(p0, h0, l0);
  split8(p0 + 8, h1, l1);
  ahi = cat8(h0, h1); alo = cat8(l0, l1);
}

// B-fragment (32x16 bf16, lane->N, halves hold K 0..15 / 16..31, matching A)
static __device__ inline void loadB_f32(const float* __restrict__ rowp, int kbase, int half,
                                        bf16v& bhi, bf16v& blo) {
  const float* p0 = rowp + kbase + half * 16;
  bf8 h0, l0, h1, l1;
  split8(p0, h0, l0);
  split8(p0 + 8, h1, l1);
  bhi = cat8(h0, h1); blo = cat8(l0, l1);
}

// ---------------------------------------------------------------------------
// Tensor Data Mover: 2D tile (64 rows x 256 B) global -> LDS with row padding
// to a 272-B LDS pitch (pad_interval=5: every 64 DWORDs, pad_amount=3: 4 DWORDs)
// so bank = 4*row -> conflict-free ds_load_b128 afterwards.
// ---------------------------------------------------------------------------
static __device__ inline void tdm_load_2d(uint32_t lds_addr, const void* gptr) {
  uint64_t ga = (uint64_t)(uintptr_t)gptr;
  u32x4 g0;
  g0[0] = 1u;                                               // count=1 (user descriptor)
  g0[1] = lds_addr;                                         // D# bits [63:32]: LDS byte addr
  g0[2] = (uint32_t)ga;                                     // global_addr[31:0]
  g0[3] = (uint32_t)((ga >> 32) & 0x1FFFFFFull) | (2u << 30); // global_addr[56:32] | type=2
  i32x8 g1;
  g1[0] = (2 << 16) | (1 << 20) | (5 << 22) | (3 << 25);    // data_size=4B, pad en, intv, amt
  g1[1] = (64 << 16);                                       // tensor_dim0 = 64 units (256 B)
  g1[2] = (64 << 16);                                       // tensor_dim1 = 64 rows
  g1[3] = (64 << 16);                                       // tile_dim0 = 64 units
  g1[4] = 64;                                               // tile_dim1 = 64 rows
  g1[5] = 512;                                              // tensor_dim0_stride = 2048 B
  g1[6] = 0;
  g1[7] = 0;
  i32x4 z4 = {0, 0, 0, 0};
#if __clang_major__ >= 23
  i32x8 z8 = {0, 0, 0, 0, 0, 0, 0, 0};
  __builtin_amdgcn_tensor_load_to_lds(g0, g1, z4, z4, z8, 0);
#else
  __builtin_amdgcn_tensor_load_to_lds(g0, g1, z4, z4, 0);
#endif
}

// ---------------------------------------------------------------------------
// Kernel 0: reset grid-barrier state and zero the t=0 bf16 state double buffer.
// ---------------------------------------------------------------------------
__global__ void rnn_init_kernel(uint32_t* __restrict__ bar, uint32_t* __restrict__ rbuf32) {
  uint32_t idx = blockIdx.x * blockDim.x + threadIdx.x;
  if (idx < (uint32_t)(2 * B_ * REC_ / 2)) rbuf32[idx] = 0u;  // Rhi0 + Rlo0
  if (blockIdx.x == 0 && threadIdx.x < 2) bar[threadIdx.x] = 0u;
}

// ---------------------------------------------------------------------------
// Kernel 1: drive[b,t,:] = inp[b,t,:]@Wi^T + bi + 0.1*noise[t] + br -> rout[b,t]
// (slot [b,0] zeroed: it must hold r0 = 0). One 16x16 tile per wave, K=128.
// ---------------------------------------------------------------------------
__global__ void __launch_bounds__(256) rnn_drive_kernel(
    const float* __restrict__ inp, const float* __restrict__ noise,
    const float* __restrict__ Wi, const float* __restrict__ bi,
    const float* __restrict__ br, float* __restrict__ rout) {
  const int wave = threadIdx.x >> 5, lane = threadIdx.x & 31;
  const int li = lane & 15, half = lane >> 4;
  const int tile = blockIdx.x * 8 + wave;   // 131072 tiles total
  const int ntile = tile & 63;              // REC/16 = 64
  const int mtile = tile >> 6;              // (B*T)/16 = 2048
  const int n0 = ntile * 16;
  const int b  = mtile >> 5;                // 512/16 = 32 m-tiles per batch row
  const int t0 = (mtile & 31) * 16;
  const int n  = n0 + li;

  v8f acc;
#pragma unroll
  for (int v = 0; v < 8; ++v) {
    int t = t0 + v + 8 * half;              // C/D layout: row = v + 8*half
    acc[v] = bi[n] + br[n] + NSD_ * noise[(size_t)t * REC_ + n];
  }
  const float* arow = inp + (size_t)(b * T_ + t0 + li) * INP_;
  const float* brow = Wi + (size_t)n * INP_;
#pragma unroll
  for (int k32 = 0; k32 < INP_ / 32; ++k32) {
    bf16v ah, al, bh, bl;
    loadA_f32(arow, k32 * 32, half, ah, al);
    loadB_f32(brow, k32 * 32, half, bh, bl);
    acc = wmma3(ah, al, bh, bl, acc);
  }
#pragma unroll
  for (int v = 0; v < 8; ++v) {
    int t = t0 + v + 8 * half;
    rout[(size_t)(b * T_ + t) * REC_ + n] = (t == 0) ? 0.0f : acc[v];
  }
}

// ---------------------------------------------------------------------------
// Kernel 2: persistent scan. 32 WGs x 256 threads. Wr slice lives in LDS as
// pre-swizzled bf16 hi/lo B-fragments; the r state (A operand) is streamed
// into LDS each step by the Tensor Data Mover, double-buffered in K-chunks.
// One grid barrier per timestep.
// ---------------------------------------------------------------------------
#define NWG_SCAN 32
#define LDS_BLOCK_STRIDE 2192                    // 32 cols * 64B + bank pad, 16B aligned
#define LDS_WR_BYTES (64 * LDS_BLOCK_STRIDE)     // 140288
#define A_ROW_PITCH   272                        // 256 B row + 16 B TDM pad
#define A_PLANE_BYTES (64 * A_ROW_PITCH)         // 17408 (one hi or lo plane chunk)
#define A_BUF_STRIDE  (2 * A_PLANE_BYTES)        // 34816 (hi + lo)
#define LDS_BYTES (LDS_WR_BYTES + 2 * A_BUF_STRIDE)  // 209920 <= 320K

static __device__ inline int lds_off(int k32, int eh, int c) {
  // Wr entry: 16 bf16 hi then 16 bf16 lo (64 B), padded against bank conflicts
  return (k32 * 2 + eh) * LDS_BLOCK_STRIDE + c * 64 + (c >> 2) * 16;
}

__global__ void __launch_bounds__(256, 1) rnn_scan_kernel(
    const float* __restrict__ Wr, float* __restrict__ rout,
    __bf16* __restrict__ rbuf, uint32_t* __restrict__ bar) {
  extern __shared__ char smem[];
  const int tid = threadIdx.x;
  const int g = blockIdx.x;
  const int wave = tid >> 5, lane = tid & 31;
  const int li = lane & 15, half = lane >> 4;
  const int ntile = wave >> 2, mtile = wave & 3;   // 2 N-tiles x 4 M-tiles per WG
  const int n0 = g * 32 + ntile * 16;
  const int m0 = mtile * 16;
  const int n = n0 + li;
  const uint32_t lds_abase = (uint32_t)(uintptr_t)smem + LDS_WR_BYTES;

  // One-time: stage this WG's 32 Wr columns into LDS, bf16 hi/lo, B-fragment order.
  for (int e = tid; e < 32 * 2 * 32; e += 256) {
    int c = e & 31, eh = (e >> 5) & 1, k32 = e >> 6;
    const float* src = Wr + (size_t)(g * 32 + c) * REC_ + k32 * 32 + eh * 16;
    __bf16* dst = (__bf16*)(smem + lds_off(k32, eh, c));
#pragma unroll
    for (int i = 0; i < 16; ++i) {
      float x = src[i];
      __bf16 h = (__bf16)x;
      dst[i] = h;
      dst[16 + i] = (__bf16)(x - (float)h);
    }
  }
  __syncthreads();

  const size_t SZ = (size_t)B_ * REC_;   // one bf16 state plane
  for (int t = 1; t < T_; ++t) {
    const __bf16* rh_src = rbuf + (size_t)((t - 1) & 1) * 2 * SZ;
    const __bf16* rl_src = rh_src + SZ;
    __bf16* rh_dst = rbuf + (size_t)(t & 1) * 2 * SZ;
    __bf16* rl_dst = rh_dst + SZ;

    // acc <- drive tile (C/D layout) already stored at slot [b, t]
    v8f acc;
#pragma unroll
    for (int v = 0; v < 8; ++v) {
      int bb = m0 + v + 8 * half;
      acc[v] = rout[(size_t)(bb * T_ + t) * REC_ + n];
    }

    // TDM pipeline: chunk c = K range [c*128, c*128+128), hi+lo planes.
    if (wave == 0) {
      tdm_load_2d(lds_abase, rh_src);                       // chunk 0 hi
      tdm_load_2d(lds_abase + A_PLANE_BYTES, rl_src);       // chunk 0 lo
    }
    for (int c = 0; c < 8; ++c) {
      if (wave == 0) {
        if (c < 7) {
          uint32_t dst = lds_abase + ((c + 1) & 1) * A_BUF_STRIDE;
          tdm_load_2d(dst, rh_src + (size_t)(c + 1) * 128);
          tdm_load_2d(dst + A_PLANE_BYTES, rl_src + (size_t)(c + 1) * 128);
          __builtin_amdgcn_s_wait_tensorcnt(2);  // chunk c landed; c+1 in flight
        } else {
          __builtin_amdgcn_s_wait_tensorcnt(0);
        }
      }
      __syncthreads();  // chunk c visible to all waves
      const char* ah_base = smem + LDS_WR_BYTES + (c & 1) * A_BUF_STRIDE
                          + (m0 + li) * A_ROW_PITCH;
      const char* al_base = ah_base + A_PLANE_BYTES;
#pragma unroll
      for (int k32l = 0; k32l < 4; ++k32l) {
        const int ko = k32l * 64 + half * 32;               // byte offset in 272-B row
        bf16v ah = cat8(*(const bf8*)(ah_base + ko), *(const bf8*)(ah_base + ko + 16));
        bf16v al = cat8(*(const bf8*)(al_base + ko), *(const bf8*)(al_base + ko + 16));
        const bf8* pb = (const bf8*)(smem + lds_off(c * 4 + k32l, half, ntile * 16 + li));
        bf16v bh = cat8(pb[0], pb[1]);
        bf16v bl = cat8(pb[2], pb[3]);
        acc = wmma3(ah, al, bh, bl, acc);
      }
      __syncthreads();  // all reads done before buffer (c&1) is overwritten
    }

    // r_new = 0.9*r_prev + 0.1*relu(h); publish fp32 to d_out and bf16 hi/lo to rbuf
#pragma unroll
    for (int v = 0; v < 8; ++v) {
      int bb = m0 + v + 8 * half;
      float rp = rout[(size_t)(bb * T_ + t - 1) * REC_ + n];
      float rn = (1.0f - ALPHA_) * rp + ALPHA_ * fmaxf(acc[v], 0.0f);
      rout[(size_t)(bb * T_ + t) * REC_ + n] = rn;
      __bf16 h = (__bf16)rn;
      rh_dst[(size_t)bb * REC_ + n] = h;
      rl_dst[(size_t)bb * REC_ + n] = (__bf16)(rn - (float)h);
      if (t + 1 < T_)  // warm L2 for next step's drive tile (global_prefetch_b8)
        __builtin_prefetch(&rout[(size_t)(bb * T_ + t + 1) * REC_ + n], 0, 1);
    }

    // ---- grid barrier (sense-reversing, device-scope release/acquire) ----
    __builtin_amdgcn_fence(__ATOMIC_RELEASE, "agent");
    __syncthreads();
    if (tid == 0) {
      uint32_t gsnap = __atomic_load_n(&bar[1], __ATOMIC_RELAXED);
      uint32_t arrived = __atomic_add_fetch(&bar[0], 1u, __ATOMIC_RELAXED);
      if (arrived == (uint32_t)gridDim.x) {
        __atomic_store_n(&bar[0], 0u, __ATOMIC_RELAXED);
        __atomic_store_n(&bar[1], gsnap + 1u, __ATOMIC_RELEASE);
      } else {
        while (__atomic_load_n(&bar[1], __ATOMIC_ACQUIRE) == gsnap)
          __builtin_amdgcn_s_sleep(1);
      }
    }
    __syncthreads();
    __builtin_amdgcn_fence(__ATOMIC_ACQUIRE, "agent");
  }
}

// ---------------------------------------------------------------------------
// Kernel 3: out[b,t,:] = r[b,t,:]@Wo^T + bo  (t>=1; out[:,0,:]=0). K=1024.
// ---------------------------------------------------------------------------
__global__ void __launch_bounds__(256) rnn_out_kernel(
    const float* __restrict__ rout, const float* __restrict__ Wo,
    const float* __restrict__ bo, float* __restrict__ out) {
  const int wave = threadIdx.x >> 5, lane = threadIdx.x & 31;
  const int li = lane & 15, half = lane >> 4;
  const int tile = blockIdx.x * 8 + wave;   // 8192 tiles
  const int ntile = tile & 3;               // OUT/16 = 4
  const int mtile = tile >> 2;              // 2048
  const int n0 = ntile * 16;
  const int b  = mtile >> 5;
  const int t0 = (mtile & 31) * 16;
  const int n  = n0 + li;

  v8f acc;
#pragma unroll
  for (int v = 0; v < 8; ++v) acc[v] = bo[n];
  const float* arow = rout + (size_t)(b * T_ + t0 + li) * REC_;
  const float* brow = Wo + (size_t)n * REC_;
#pragma unroll 4
  for (int k32 = 0; k32 < REC_ / 32; ++k32) {
    bf16v ah, al, bh, bl;
    loadA_f32(arow, k32 * 32, half, ah, al);
    loadB_f32(brow, k32 * 32, half, bh, bl);
    acc = wmma3(ah, al, bh, bl, acc);
  }
#pragma unroll
  for (int v = 0; v < 8; ++v) {
    int t = t0 + v + 8 * half;
    out[(size_t)(b * T_ + t) * OUT_ + n] = (t == 0) ? 0.0f : acc[v];
  }
}

// ---------------------------------------------------------------------------
extern "C" void kernel_launch(void* const* d_in, const int* in_sizes, int n_in,
                              void* d_out, int out_size, void* d_ws, size_t ws_size,
                              hipStream_t stream) {
  (void)in_sizes; (void)n_in; (void)out_size; (void)ws_size;
  const float* inp   = (const float*)d_in[0];
  const float* noise = (const float*)d_in[1];
  const float* Wi    = (const float*)d_in[2];
  const float* bi    = (const float*)d_in[3];
  const float* Wr    = (const float*)d_in[4];
  const float* br    = (const float*)d_in[5];
  const float* Wo    = (const float*)d_in[6];
  const float* bo    = (const float*)d_in[7];

  float* out  = (float*)d_out;
  float* rout = out + (size_t)B_ * T_ * OUT_;   // r region of d_out (drive stored in-place)

  uint32_t* bar = (uint32_t*)d_ws;                    // [0]=count, [1]=generation
  __bf16*   rbuf = (__bf16*)((char*)d_ws + 256);      // 2x (Rhi,Rlo) double buffer, 512 KB

  (void)hipFuncSetAttribute(reinterpret_cast<const void*>(rnn_scan_kernel),
                            hipFuncAttributeMaxDynamicSharedMemorySize, LDS_BYTES);

  rnn_init_kernel <<<256,   256, 0, stream>>>(bar, (uint32_t*)rbuf);
  rnn_drive_kernel<<<16384, 256, 0, stream>>>(inp, noise, Wi, bi, br, rout);
  rnn_scan_kernel <<<NWG_SCAN, 256, LDS_BYTES, stream>>>(Wr, rout, rbuf, bar);
  rnn_out_kernel  <<<1024,  256, 0, stream>>>(rout, Wo, bo, out);
}
